// Decoder_34634616275454
// MI455X (gfx1250) — matrix operand order repaired and checked by reading, not compiled, not run on hardware
//
#include <hip/hip_runtime.h>
#include <hip/hip_bf16.h>

#define BATCH 512
#define TSEQ  256
#define EDIM  256
#define DDIM  256

typedef __attribute__((ext_vector_type(16))) __bf16 v16bf;
typedef __attribute__((ext_vector_type(8)))  __bf16 v8bf;
typedef __attribute__((ext_vector_type(4)))  __bf16 v4bf;
typedef __attribute__((ext_vector_type(8)))  float  v8f;

__device__ __forceinline__ float fexp2(float x){ return __builtin_amdgcn_exp2f(x); }
__device__ __forceinline__ float frcpf_(float x){ return __builtin_amdgcn_rcpf(x); }
__device__ __forceinline__ float fsigmoid(float x){ return frcpf_(1.0f + fexp2(-1.44269504f * x)); }

#if defined(__has_builtin)
#  if __has_builtin(__builtin_amdgcn_tanhf)
#    define FTANH_NATIVE 1
__device__ __forceinline__ float ftanh_(float x){ return __builtin_amdgcn_tanhf(x); }
#  elif __has_builtin(__builtin_amdgcn_tanh_f32)
#    define FTANH_NATIVE 1
__device__ __forceinline__ float ftanh_(float x){ return __builtin_amdgcn_tanh_f32(x); }
#  endif
#endif
#ifndef FTANH_NATIVE
__device__ __forceinline__ float ftanh_(float x){ return 2.0f * frcpf_(1.0f + fexp2(-2.885390082f * x)) - 1.0f; }
#endif

__device__ __forceinline__ float wsum(float v){
  #pragma unroll
  for (int m = 16; m > 0; m >>= 1) v += __shfl_xor(v, m, 32);
  return v;
}
__device__ __forceinline__ float wmax(float v){
  #pragma unroll
  for (int m = 16; m > 0; m >>= 1) v = fmaxf(v, __shfl_xor(v, m, 32));
  return v;
}

// ---------------- Kernel 1: f32 -> bf16 convert (input_encoded) ----------------
__global__ void cvt_input_kernel(const float* __restrict__ src, __bf16* __restrict__ dst, int n4)
{
  int i = blockIdx.x * blockDim.x + threadIdx.x;
  if (i >= n4) return;
  float4 f = ((const float4*)src)[i];
  v4bf o;
  o[0] = (__bf16)f.x; o[1] = (__bf16)f.y; o[2] = (__bf16)f.z; o[3] = (__bf16)f.w;
  ((v4bf*)dst)[i] = o;
}

// ---------------- Kernel 2: weight repack to bf16 ----------------
// attn_w1 (256 x 1024): cols [0,512) -> w1_hc (256x512), cols [512,768) -> w1_enc (256x256)
// W_hh (1024 x 256): direct convert
__global__ void cvt_weights_kernel(const float* __restrict__ attn_w1, const float* __restrict__ W_hh,
                                   __bf16* __restrict__ w1_encB, __bf16* __restrict__ w1_hcB,
                                   __bf16* __restrict__ W_hhB)
{
  int idx = blockIdx.x * blockDim.x + threadIdx.x;
  if (idx < 256 * 256) {
    int f = idx >> 8, e = idx & 255;
    w1_encB[idx] = (__bf16)attn_w1[(size_t)f * 1024 + 512 + e];
  } else if (idx < 256 * 256 + 256 * 512) {
    int k2 = idx - 256 * 256;
    int f = k2 >> 9, k = k2 & 511;
    w1_hcB[k2] = (__bf16)attn_w1[(size_t)f * 1024 + k];
  } else if (idx < 256 * 256 + 256 * 512 + 1024 * 256) {
    int k3 = idx - (256 * 256 + 256 * 512);
    W_hhB[k3] = (__bf16)W_hh[k3];
  }
}

// ---------------- Kernel 3: enc_proj = input @ w1_enc^T + b1 (WMMA bf16) ----------------
// M = B*T = 131072, N = 256, K = 256. One wave per 16x16 output tile; 8 waves/block.
__global__ __launch_bounds__(256) void encproj_kernel(
    const __bf16* __restrict__ inputB, const __bf16* __restrict__ w1_encB,
    const float* __restrict__ attn_b1, __bf16* __restrict__ encB)
{
  const int tid = threadIdx.x, waveId = tid >> 5, lane = tid & 31;
  const int lane16 = lane & 15, hi = lane >> 4;
  const int tile  = blockIdx.x * 8 + waveId;
  const int mtile = tile >> 4, ntile = tile & 15;
  const size_t arow = (size_t)(mtile * 16 + lane16) * EDIM;
  const size_t brow = (size_t)(ntile * 16 + lane16) * EDIM;

  v8f acc = {};
  #pragma unroll 1
  for (int kk = 0; kk < EDIM; kk += 32) {
    const int kb = kk + hi * 16;
    const v8bf* ap = (const v8bf*)(inputB + arow + kb);
    v8bf a0 = ap[0], a1 = ap[1];          // K = kb..kb+7  and  kb+8..kb+15
    v16bf a;
    #pragma unroll
    for (int i = 0; i < 8; ++i) { a[i] = a0[i]; a[8 + i] = a1[i]; }
    v16bf bb = *(const v16bf*)(w1_encB + brow + kk + hi * 16);
    acc = __builtin_amdgcn_wmma_f32_16x16x32_bf16(false, a, false, bb, (short)0, acc, false, false);
  }
  const int n = ntile * 16 + lane16;
  const float bias = attn_b1[n];
  #pragma unroll
  for (int r = 0; r < 8; ++r)
    encB[(size_t)(mtile * 16 + r + hi * 8) * EDIM + n] = (__bf16)(acc[r] + bias);
}

// ---------------- Kernel 4: fused persistent decoder ----------------
// One block (16 waves) owns 16 batch rows for all 256 recurrent steps.
// h and c kept as bf16 rows in LDS so WMMA A-fragments are direct ds_load_b128 pairs.
#define HB_STRIDE 264   // 8-element pad: row stride = 132 dwords == 4 mod 64 banks

__global__ __launch_bounds__(512, 1) void decoder_kernel(
    const __bf16* __restrict__ inputB, const __bf16* __restrict__ encB,
    const __bf16* __restrict__ w1_hcB, const __bf16* __restrict__ W_hhB,
    const float* __restrict__ attn_w2, const float* __restrict__ attn_b2,
    const float* __restrict__ y_hist,
    const float* __restrict__ W_ih, const float* __restrict__ b_ih, const float* __restrict__ b_hh,
    const float* __restrict__ fc_w, const float* __restrict__ fc_b,
    const float* __restrict__ fcf_w, const float* __restrict__ fcf_b,
    float* __restrict__ out)
{
  const int tid = threadIdx.x;
  const int waveId = tid >> 5;
  const int lane = tid & 31;
  const int lane16 = lane & 15;
  const int hi = lane >> 4;
  const int b0 = blockIdx.x * 16;

  __shared__ float  sh_c[16][256];          // f32 cell state (accumulates across steps)
  __shared__ __bf16 sh_hb[16][HB_STRIDE];   // bf16 h, WMMA-feed layout
  __shared__ __bf16 sh_cb[16][HB_STRIDE];   // bf16 c, WMMA-feed layout
  __shared__ float  sh_proj[16][256];
  __shared__ float  sh_attn[16][256];
  __shared__ float  sh_ctx[16][256];
  __shared__ float  sh_gates[16][1024];
  __shared__ float  sh_w2[256];
  __shared__ float  sh_yt[16];

  for (int i = tid; i < 16 * HB_STRIDE; i += 512) {
    (&sh_hb[0][0])[i] = (__bf16)0.f;
    (&sh_cb[0][0])[i] = (__bf16)0.f;
  }
  for (int i = tid; i < 16 * 256; i += 512) {
    (&sh_c[0][0])[i] = 0.f;
    (&sh_ctx[0][0])[i] = 0.f;
  }
  if (tid < 256) sh_w2[tid] = attn_w2[tid];
  const float b2    = attn_b2[0];
  const float fcb   = fc_b[0];
  const float fcw_y = fc_w[EDIM];
  __syncthreads();

  #pragma unroll 1
  for (int t = 0; t < TSEQ; ++t) {
    // ---- 1) proj = [h|c] @ w1_hc^T : 16x256 out, one 16x16 n-tile per wave ----
    {
      const int n = waveId * 16 + lane16;
      v8f acc = {};
      #pragma unroll 1
      for (int kk = 0; kk < 2 * DDIM; kk += 32) {
        const __bf16* srcb = (kk < DDIM) ? &sh_hb[lane16][0] : &sh_cb[lane16][0];
        const int ko = (kk & (DDIM - 1)) + hi * 16;
        v8bf a0 = *(const v8bf*)(srcb + ko);
        v8bf a1 = *(const v8bf*)(srcb + ko + 8);
        v16bf a;
        #pragma unroll
        for (int i = 0; i < 8; ++i) { a[i] = a0[i]; a[8 + i] = a1[i]; }
        v16bf bb = *(const v16bf*)(w1_hcB + (size_t)n * (2 * DDIM) + kk + hi * 16);
        acc = __builtin_amdgcn_wmma_f32_16x16x32_bf16(false, a, false, bb, (short)0, acc, false, false);
      }
      #pragma unroll
      for (int r = 0; r < 8; ++r) sh_proj[r + hi * 8][n] = acc[r];
    }
    __syncthreads();

    // ---- 2) scores[b][tt] = sum_e tanh(encp + proj) * w2  (L2-resident bf16 stream) ----
    #pragma unroll 1
    for (int it = 0; it < 8; ++it) {
      const int idx = it * 512 + tid;
      const int b = idx >> 8, tt = idx & 255;
      const v8bf* ep = (const v8bf*)(encB + ((size_t)(b0 + b) * TSEQ + tt) * EDIM);
      float acc = b2;
      #pragma unroll 4
      for (int c8 = 0; c8 < EDIM / 8; ++c8) {
        v8bf ev = ep[c8];
        #pragma unroll
        for (int j2 = 0; j2 < 8; ++j2) {
          const int e = c8 * 8 + j2;
          acc += ftanh_((float)ev[j2] + sh_proj[b][e]) * sh_w2[e];
        }
      }
      sh_attn[b][tt] = acc;
    }
    __syncthreads();

    // ---- 3) softmax over tt, one wave per row ----
    {
      const int b = waveId;
      float v[8];
      float m = -3.0e38f;
      #pragma unroll
      for (int i = 0; i < 8; ++i) { v[i] = sh_attn[b][lane + 32 * i]; m = fmaxf(m, v[i]); }
      m = wmax(m);
      float s = 0.f;
      #pragma unroll
      for (int i = 0; i < 8; ++i) { v[i] = fexp2(1.44269504f * (v[i] - m)); s += v[i]; }
      s = wsum(s);
      const float inv = frcpf_(s);
      #pragma unroll
      for (int i = 0; i < 8; ++i) sh_attn[b][lane + 32 * i] = v[i] * inv;
    }
    __syncthreads();

    // ---- 4) context[b][e] = sum_tt alpha * input (coalesced bf16 reads) ----
    #pragma unroll 1
    for (int it = 0; it < 8; ++it) {
      const int idx = it * 512 + tid;
      const int b = idx >> 8, e = idx & 255;
      const __bf16* ip = inputB + ((size_t)(b0 + b) * TSEQ) * EDIM + e;
      float acc = 0.f;
      #pragma unroll 4
      for (int tt = 0; tt < TSEQ; ++tt)
        acc += sh_attn[b][tt] * (float)ip[(size_t)tt * EDIM];
      sh_ctx[b][e] = acc;
    }
    __syncthreads();

    // ---- y_tilde[b] = dot(fc_w[0:256], ctx) + fc_w[256]*y_t + fc_b ----
    {
      const int b = waveId;
      float acc = 0.f;
      #pragma unroll
      for (int i = 0; i < 8; ++i) { const int e = lane + 32 * i; acc += sh_ctx[b][e] * fc_w[e]; }
      acc = wsum(acc);
      if (lane == 0)
        sh_yt[b] = acc + fcw_y * y_hist[(size_t)(b0 + b) * TSEQ + t] + fcb;
    }
    __syncthreads();

    // ---- 5) gates = h @ W_hh^T + y_tilde*W_ih + b_ih + b_hh : 4 n-tiles per wave ----
    #pragma unroll 1
    for (int j = 0; j < 4; ++j) {
      const int n = (waveId * 4 + j) * 16 + lane16;
      v8f acc = {};
      #pragma unroll 1
      for (int kk = 0; kk < DDIM; kk += 32) {
        const __bf16* srcb = &sh_hb[lane16][0];
        const int ko = kk + hi * 16;
        v8bf a0 = *(const v8bf*)(srcb + ko);
        v8bf a1 = *(const v8bf*)(srcb + ko + 8);
        v16bf a;
        #pragma unroll
        for (int i = 0; i < 8; ++i) { a[i] = a0[i]; a[8 + i] = a1[i]; }
        v16bf bb = *(const v16bf*)(W_hhB + (size_t)n * DDIM + kk + hi * 16);
        acc = __builtin_amdgcn_wmma_f32_16x16x32_bf16(false, a, false, bb, (short)0, acc, false, false);
      }
      const float wih  = W_ih[n];
      const float bias = b_ih[n] + b_hh[n];
      #pragma unroll
      for (int r = 0; r < 8; ++r) {
        const int b = r + hi * 8;
        sh_gates[b][n] = acc[r] + sh_yt[b] * wih + bias;
      }
    }
    __syncthreads();

    // ---- 6) LSTM pointwise update (writes f32 c + bf16 h/c copies) ----
    #pragma unroll 1
    for (int it = 0; it < 8; ++it) {
      const int idx = it * 512 + tid;
      const int b = idx >> 8, d = idx & 255;
      const float ig = fsigmoid(sh_gates[b][d]);
      const float fg = fsigmoid(sh_gates[b][DDIM + d]);
      const float gg = ftanh_(sh_gates[b][2 * DDIM + d]);
      const float og = fsigmoid(sh_gates[b][3 * DDIM + d]);
      const float cn = fg * sh_c[b][d] + ig * gg;
      const float hn = og * ftanh_(cn);
      sh_c[b][d]  = cn;
      sh_cb[b][d] = (__bf16)cn;
      sh_hb[b][d] = (__bf16)hn;
    }
    __syncthreads();
  }

  // ---- final: out = [h | ctx] @ fcf_w^T + fcf_b, one wave per row ----
  {
    const int b = waveId;
    #pragma unroll
    for (int oo = 0; oo < 2; ++oo) {
      float acc = 0.f;
      #pragma unroll
      for (int i = 0; i < 16; ++i) {
        const int k = lane + 32 * i;
        const float x = (k < DDIM) ? (float)sh_hb[b][k] : sh_ctx[b][k - DDIM];
        acc += x * fcf_w[oo * (DDIM + EDIM) + k];
      }
      acc = wsum(acc);
      if (lane == 0) out[(size_t)(b0 + b) * 2 + oo] = acc + fcf_b[oo];
    }
  }
}

extern "C" void kernel_launch(void* const* d_in, const int* in_sizes, int n_in,
                              void* d_out, int out_size, void* d_ws, size_t ws_size,
                              hipStream_t stream)
{
  const float* input_encoded = (const float*)d_in[0];
  const float* y_history     = (const float*)d_in[1];
  const float* attn_w1       = (const float*)d_in[2];
  const float* attn_b1       = (const float*)d_in[3];
  const float* attn_w2       = (const float*)d_in[4];
  const float* attn_b2       = (const float*)d_in[5];
  const float* W_ih          = (const float*)d_in[6];
  const float* W_hh          = (const float*)d_in[7];
  const float* b_ih          = (const float*)d_in[8];
  const float* b_hh          = (const float*)d_in[9];
  const float* fc_w          = (const float*)d_in[10];
  const float* fc_b          = (const float*)d_in[11];
  const float* fcf_w         = (const float*)d_in[12];
  const float* fcf_b         = (const float*)d_in[13];
  float* out = (float*)d_out;

  const size_t nBTE = (size_t)BATCH * TSEQ * EDIM;   // 33,554,432
  __bf16* inputB  = (__bf16*)d_ws;
  __bf16* encB    = inputB + nBTE;
  __bf16* w1_encB = encB + nBTE;
  __bf16* w1_hcB  = w1_encB + 256 * 256;
  __bf16* W_hhB   = w1_hcB + 256 * 512;

  cvt_input_kernel<<<(int)(nBTE / 4 / 256), 256, 0, stream>>>(input_encoded, inputB, (int)(nBTE / 4));

  const int wn = 256 * 256 + 256 * 512 + 1024 * 256;
  cvt_weights_kernel<<<(wn + 255) / 256, 256, 0, stream>>>(attn_w1, W_hh, w1_encB, w1_hcB, W_hhB);

  encproj_kernel<<<(BATCH * TSEQ / 16) * (EDIM / 16) / 8, 256, 0, stream>>>(inputB, w1_encB, attn_b1, encB);

  decoder_kernel<<<BATCH / 16, 512, 0, stream>>>(inputB, encB, w1_hcB, W_hhB,
                                                 attn_w2, attn_b2, y_history,
                                                 W_ih, b_ih, b_hh, fc_w, fc_b, fcf_w, fcf_b, out);
}
